// NMILoss_15161234555310
// MI455X (gfx1250) — matrix operand order repaired and checked by reading, not compiled, not run on hardware
//
#include <hip/hip_runtime.h>

typedef __attribute__((ext_vector_type(16))) _Float16 v16h;
typedef __attribute__((ext_vector_type(8)))  float    v8f;
typedef __attribute__((ext_vector_type(4)))  unsigned int v4u;

#define NBINS 64
#define RBLOCKS 256
#define KTILE 128
#define VOX_PER_BLOCK 2048

// ---------------- pass 1: per-block min/max of target & source ----------------
__global__ __launch_bounds__(256) void nmi_minmax1(const float* __restrict__ t,
                                                   const float* __restrict__ s,
                                                   float* __restrict__ part,
                                                   int total) {
  __shared__ float red[256];
  const int tid = threadIdx.x;
  float mnT = __FLT_MAX__, mxT = -__FLT_MAX__;
  float mnS = __FLT_MAX__, mxS = -__FLT_MAX__;
  for (int i = blockIdx.x * blockDim.x + tid; i < total; i += gridDim.x * blockDim.x) {
    float a = t[i]; mnT = fminf(mnT, a); mxT = fmaxf(mxT, a);
    float c = s[i]; mnS = fminf(mnS, c); mxS = fmaxf(mxS, c);
  }
  red[tid] = mnT; __syncthreads();
  for (int o = 128; o > 0; o >>= 1) { if (tid < o) red[tid] = fminf(red[tid], red[tid + o]); __syncthreads(); }
  if (tid == 0) part[0 * RBLOCKS + blockIdx.x] = red[0];
  __syncthreads();
  red[tid] = mxT; __syncthreads();
  for (int o = 128; o > 0; o >>= 1) { if (tid < o) red[tid] = fmaxf(red[tid], red[tid + o]); __syncthreads(); }
  if (tid == 0) part[1 * RBLOCKS + blockIdx.x] = red[0];
  __syncthreads();
  red[tid] = mnS; __syncthreads();
  for (int o = 128; o > 0; o >>= 1) { if (tid < o) red[tid] = fminf(red[tid], red[tid + o]); __syncthreads(); }
  if (tid == 0) part[2 * RBLOCKS + blockIdx.x] = red[0];
  __syncthreads();
  red[tid] = mxS; __syncthreads();
  for (int o = 128; o > 0; o >>= 1) { if (tid < o) red[tid] = fmaxf(red[tid], red[tid + o]); __syncthreads(); }
  if (tid == 0) part[3 * RBLOCKS + blockIdx.x] = red[0];
}

// -------- pass 2: finish min/max -> (min, 64/range); also zero the histogram --------
__global__ __launch_bounds__(256) void nmi_minmax2(const float* __restrict__ part,
                                                   float* __restrict__ mm,
                                                   float* __restrict__ hist) {
  __shared__ float red[256];
  const int tid = threadIdx.x;
  for (int i = tid; i < 2 * NBINS * NBINS; i += 256) hist[i] = 0.0f;

  float mnT, mxT, mnS, mxS;
  red[tid] = part[0 * RBLOCKS + tid]; __syncthreads();
  for (int o = 128; o > 0; o >>= 1) { if (tid < o) red[tid] = fminf(red[tid], red[tid + o]); __syncthreads(); }
  mnT = red[0]; __syncthreads();
  red[tid] = part[1 * RBLOCKS + tid]; __syncthreads();
  for (int o = 128; o > 0; o >>= 1) { if (tid < o) red[tid] = fmaxf(red[tid], red[tid + o]); __syncthreads(); }
  mxT = red[0]; __syncthreads();
  red[tid] = part[2 * RBLOCKS + tid]; __syncthreads();
  for (int o = 128; o > 0; o >>= 1) { if (tid < o) red[tid] = fminf(red[tid], red[tid + o]); __syncthreads(); }
  mnS = red[0]; __syncthreads();
  red[tid] = part[3 * RBLOCKS + tid]; __syncthreads();
  for (int o = 128; o > 0; o >>= 1) { if (tid < o) red[tid] = fmaxf(red[tid], red[tid + o]); __syncthreads(); }
  mxS = red[0];
  if (tid == 0) {
    mm[0] = mnT; mm[1] = (float)NBINS / (mxT - mnT);
    mm[2] = mnS; mm[3] = (float)NBINS / (mxS - mnS);
  }
}

// ---------------- joint soft-histogram via WMMA: hist[b,64,64] += Wt * Ws^T ----------------
// Per 128-voxel fill: zero the 64x128 f16 W tiles with b128 stores, scatter the
// 4 non-zero B-spline weights per voxel, then run 4 unrolled 16x16x32 WMMA K-steps.
// Voxel loads are software-pipelined one fill ahead so the s_wait_loadcnt lands
// behind the WMMA phase; prefetch covers the fill after that.
__global__ __launch_bounds__(256) void nmi_hist_wmma(const float* __restrict__ t,
                                                     const float* __restrict__ s,
                                                     const float* __restrict__ mm,
                                                     float* __restrict__ hist,
                                                     int nPerBatch, int voxPerBlock) {
  __shared__ __align__(16) _Float16 W[2 * NBINS * KTILE];   // [mat][bin][k] : 32 KB

  const int tid    = threadIdx.x;
  const int b      = blockIdx.y;
  const int chunk0 = blockIdx.x * voxPerBlock;              // voxel offset inside batch
  const float mnT = mm[0], scT = mm[1], mnS = mm[2], scS = mm[3];

  const int wave = tid >> 5;
  const int lane = tid & 31;
  const int mt   = wave >> 1;                // A row-tile  (0..3)
  const int nt0  = (wave & 1) * 2;           // B col-tiles {0,1} or {2,3}
  const int nt1  = nt0 + 1;
  const int half = lane >> 4;                // lane group 0/1
  const int lc   = lane & 15;

  _Float16* Wt = W;
  _Float16* Ws = W + NBINS * KTILE;

  // scatter-task assignment: thread -> (matrix, voxel-in-tile)
  const int    mat = tid >> 7;               // 0 = target, 1 = source
  const int    kv  = tid & 127;
  const float  mn  = mat ? mnS : mnT;
  const float  sc  = mat ? scS : scT;
  const float* src = mat ? s : t;
  const size_t bBase = (size_t)b * nPerBatch;

  v8f acc0 = {};
  v8f acc1 = {};

  const int fills = voxPerBlock / KTILE;

  // pipeline prologue: branchless (address-clamped) load of fill 0
  int  gk  = chunk0 + kv;
  bool vld = (gk < nPerBatch);
  float raw = src[bBase + (vld ? gk : 0)];

  for (int fl = 0; fl < fills; ++fl) {
    __syncthreads();                          // previous fill's fragment reads done
    // 1) zero both W tiles (8 x b128 LDS stores per thread)
    {
      v4u* wz = (v4u*)W;
      v4u  z  = {0u, 0u, 0u, 0u};
#pragma unroll
      for (int j = 0; j < 8; ++j) wz[tid + 256 * j] = z;
    }
    // 2) issue NEXT fill's voxel load now; its wait hides behind this fill's WMMAs
    const int  gkN  = chunk0 + (fl + 1) * KTILE + kv;
    const bool vldN = (fl + 1 < fills) && (gkN < nPerBatch);
    const float rawN = src[bBase + (vldN ? gkN : 0)];
    {                                         // prefetch two fills ahead into L2/L0
      const int gp = gkN + KTILE;
      if (gp < nPerBatch) __builtin_prefetch(&src[bBase + gp], 0, 3);
    }
    __syncthreads();
    // 3) scatter the 4 non-zero weights of this fill's voxel (branchless B-spline)
    {
      const float u  = vld ? (raw - mn) * sc : -1.0e9f;   // tail -> no bins touched
      const int   f0 = (int)__builtin_floorf(u);
      _Float16* Wm = W + mat * (NBINS * KTILE) + kv;
#pragma unroll
      for (int d = -1; d <= 2; ++d) {
        const int bin = f0 + d;
        if (bin >= 0 && bin < NBINS) {
          const float ax = __builtin_fabsf((float)bin - u);
          const float c2 = 2.0f - ax;                         // >= 0 within support
          const float c1 = __builtin_fmaxf(1.0f - ax, 0.0f);
          const float w  = (c2 * c2 * c2 - 4.0f * c1 * c1 * c1) * (1.0f / 6.0f);
          Wm[bin * KTILE] = (_Float16)w;
        }
      }
    }
    __syncthreads();
    // 4) dense 64x64 += 64x128 * 128x64 via 4 unrolled WMMA K-steps per wave-tile
#pragma unroll
    for (int kt = 0; kt < 4; ++kt) {
      const int kk = kt * 32;
      v16h a, b0, b1;
#pragma unroll
      for (int e = 0; e < 16; ++e) {
        const int ka = (e < 8) ? ((half ? 8 : 0) + e) : ((half ? 24 : 16) + (e - 8));
        a[e] = Wt[(mt * 16 + lc) * KTILE + kk + ka];
        const int kb = ka;
        b0[e] = Ws[(nt0 * 16 + lc) * KTILE + kk + kb];
        b1[e] = Ws[(nt1 * 16 + lc) * KTILE + kk + kb];
      }
      acc0 = __builtin_amdgcn_wmma_f32_16x16x32_f16(false, a, false, b0, (short)0, acc0, false, false);
      acc1 = __builtin_amdgcn_wmma_f32_16x16x32_f16(false, a, false, b1, (short)0, acc1, false, false);
    }
    raw = rawN;
    vld = vldN;
  }

  // Merge: each (mt,nt) tile owned by exactly one wave -> 4096 atomics per block (minimal)
#pragma unroll
  for (int r = 0; r < 8; ++r) {
    const int row = mt * 16 + half * 8 + r;
    atomicAdd(&hist[((size_t)b * NBINS + row) * NBINS + nt0 * 16 + lc], acc0[r]);
    atomicAdd(&hist[((size_t)b * NBINS + row) * NBINS + nt1 * 16 + lc], acc1[r]);
  }
}

// ---------------- epilogue: entropies + NMI loss (single block) ----------------
__global__ __launch_bounds__(256) void nmi_finalize(const float* __restrict__ hist,
                                                    float* __restrict__ out) {
  __shared__ float H[2 * NBINS * NBINS];   // 32 KB
  __shared__ float red[256];
  __shared__ float acc[6];                 // {sumT,sumS,sumJ} per batch (Σ p log p)
  const int tid = threadIdx.x;
  if (tid < 6) acc[tid] = 0.0f;

  float ssum = 0.0f;
  for (int i = tid; i < 2 * NBINS * NBINS; i += 256) { float v = hist[i]; H[i] = v; ssum += v; }
  red[tid] = ssum; __syncthreads();
  for (int o = 128; o > 0; o >>= 1) { if (tid < o) red[tid] += red[tid + o]; __syncthreads(); }
  const float inv = 1.0f / red[0];

  float hj0 = 0.0f, hj1 = 0.0f;
  for (int i = tid; i < 2 * NBINS * NBINS; i += 256) {
    float p = H[i] * inv;
    float term = p * __logf(p + 1e-12f);
    if (i < NBINS * NBINS) hj0 += term; else hj1 += term;
  }
  atomicAdd(&acc[2], hj0);
  atomicAdd(&acc[5], hj1);

  if (tid < 128) {                         // rows -> p_t
    const int bb = tid >> 6, r = tid & 63;
    float ps = 0.0f;
    for (int j = 0; j < NBINS; ++j) ps += H[(bb * NBINS + r) * NBINS + j];
    ps *= inv;
    atomicAdd(&acc[bb * 3 + 0], ps * __logf(ps + 1e-12f));
  } else {                                 // cols -> p_s
    const int u = tid - 128;
    const int bb = u >> 6, c = u & 63;
    float ps = 0.0f;
    for (int i = 0; i < NBINS; ++i) ps += H[(bb * NBINS + i) * NBINS + c];
    ps *= inv;
    atomicAdd(&acc[bb * 3 + 1], ps * __logf(ps + 1e-12f));
  }
  __syncthreads();
  if (tid == 0) {
    float r0 = (acc[0] + acc[1]) / acc[2];  // H = -Σ p log p; signs cancel in ratio
    float r1 = (acc[3] + acc[4]) / acc[5];
    out[0] = -0.5f * (r0 + r1);
  }
}

extern "C" void kernel_launch(void* const* d_in, const int* in_sizes, int n_in,
                              void* d_out, int out_size, void* d_ws, size_t ws_size,
                              hipStream_t stream) {
  const float* t = (const float*)d_in[0];
  const float* s = (const float*)d_in[1];
  float* out = (float*)d_out;
  float* ws  = (float*)d_ws;

  const int total = in_sizes[0];        // 2*1*96^3
  const int nPerBatch = total / 2;

  float* mm   = ws;                     // 4 floats: minT, 64/rangeT, minS, 64/rangeS
  float* part = ws + 4;                 // 4 * RBLOCKS floats
  float* hist = ws + 2048;              // 2*64*64 floats

  nmi_minmax1<<<RBLOCKS, 256, 0, stream>>>(t, s, part, total);
  nmi_minmax2<<<1, 256, 0, stream>>>(part, mm, hist);

  const int nBlocks = (nPerBatch + VOX_PER_BLOCK - 1) / VOX_PER_BLOCK;  // 432 for 96^3
  dim3 grid(nBlocks, 2, 1);
  nmi_hist_wmma<<<grid, 256, 0, stream>>>(t, s, mm, hist, nPerBatch, VOX_PER_BLOCK);

  nmi_finalize<<<1, 256, 0, stream>>>(hist, out);
}
